// LSTMMem_27058293965476
// MI455X (gfx1250) — matrix operand order repaired and checked by reading, not compiled, not run on hardware
//
#include <hip/hip_runtime.h>

// ---------- types ----------
typedef __attribute__((ext_vector_type(16))) __bf16        v16bf;
typedef __attribute__((ext_vector_type(8)))  float         v8f;
typedef __attribute__((ext_vector_type(4)))  unsigned int  u32x4;
typedef __attribute__((ext_vector_type(4)))  float         f32x4;
typedef __attribute__((ext_vector_type(8)))  int           i32x8;
typedef __attribute__((ext_vector_type(4)))  int           i32x4;

struct Bf16x16Bits { u32x4 lo; u32x4 hi; };   // 32B, trivially copyable -> bit_cast to v16bf

// ---------- problem dims ----------
#define BATCH  256
#define TSTEPS 512
#define IDIM   256
#define DDIM   256
#define NGATE  1024           // 4*D
#define KDIM   512            // I + D  (A = [x_t | h])
#define KTILES (KDIM / 32)    // 16 k-steps of WMMA k=32
#define NTILES (NGATE / 16)   // 64 n-tiles of 16

#if __has_builtin(__builtin_amdgcn_tensor_load_to_lds) && __has_builtin(__builtin_amdgcn_s_wait_tensorcnt)
#define USE_TDM 1
#else
#define USE_TDM 0
#endif

// ---------- helpers ----------
__device__ __forceinline__ unsigned short f32_to_bf16(float f) {
    unsigned int u = __float_as_uint(f);
    unsigned int r = u + 0x7FFFu + ((u >> 16) & 1u);   // round-to-nearest-even
    return (unsigned short)(r >> 16);
}

__device__ __forceinline__ float hsig(float z) {       // Keras-1 hard_sigmoid
    return fminf(fmaxf(fmaf(0.2f, z, 0.5f), 0.0f), 1.0f);
}

__device__ __forceinline__ float tanh_f(float x) {
#if __has_builtin(__builtin_amdgcn_tanhf)
    return __builtin_amdgcn_tanhf(x);                  // v_tanh_f32 on gfx1250
#else
    return tanhf(x);
#endif
}

__device__ __forceinline__ v16bf load_bfrag_global(const unsigned short* __restrict__ wp,
                                                   int kt, int nt, int lane) {
    const u32x4* p = (const u32x4*)wp + ((size_t)((kt * NTILES + nt) * 32 + lane)) * 2;
    Bf16x16Bits bits;
    bits.lo = p[0];
    bits.hi = p[1];
    return __builtin_bit_cast(v16bf, bits);
}

#if USE_TDM
// Issue a TDM 2D tile load: 16 rows x 256 f32 of x_t -> LDS staging.
// D# per CDNA5 ISA 8.3/8.4: group0 = {count/type/addr}, group1 = {dims/tile}.
// 6-arg builtin form (clang-23 / therock-10.0): groups 2/3 + spare group + cpol.
__device__ __forceinline__ void tdm_load_xtile(const float* gptr, unsigned lds_off) {
    unsigned long long ga = (unsigned long long)(uintptr_t)gptr;
    u32x4 g0;
    g0.x = 1u;                                          // count=1, user descriptor
    g0.y = lds_off;                                     // LDS byte address of staging
    g0.z = (unsigned)(ga & 0xFFFFFFFFu);                // global_addr[31:0]
    g0.w = (unsigned)((ga >> 32) & 0x01FFFFFFu) | (2u << 30);  // addr[56:32] | type=2
    i32x8 g1;
    g1[0] = 0x00020000;                                 // wg_mask=0, data_size=2 (4B)
    g1[1] = (int)((unsigned)(IDIM & 0xFFFF) << 16);     // tensor_dim0[15:0]=256
    g1[2] = (int)((unsigned)16 << 16);                  // tensor_dim0 hi=0 | tensor_dim1=16
    g1[3] = (int)((unsigned)IDIM << 16);                // tensor_dim1 hi=0 | tile_dim0=256
    g1[4] = 16;                                         // tile_dim1=16, tile_dim2=0
    g1[5] = (int)(TSTEPS * IDIM);                       // tensor_dim0_stride lo32 = 131072
    g1[6] = 0;                                          // stride hi | dim1_stride lo (unused, 2D)
    g1[7] = 0;
    i32x4 z4 = {0, 0, 0, 0};                            // groups 2/3 unused (<=2D)
    i32x8 z8 = {0, 0, 0, 0, 0, 0, 0, 0};
    __builtin_amdgcn_tensor_load_to_lds(g0, g1, z4, z4, z8, 0);
}
#endif

// =====================================================================
// Kernel 0: pack [W;U] (fp32, row-major KDIM x NGATE) into bf16 WMMA-B
// fragment order. Packed element p = ((kt*64+nt)*32+lane)*16 + e holds
// B[k = kt*32 + (lane>>4)*16 + e][col = nt*16 + (lane&15)].
// =====================================================================
__global__ void lstm_pack_weights(const float* __restrict__ W,
                                  const float* __restrict__ U,
                                  unsigned short* __restrict__ wpack) {
    int p = blockIdx.x * blockDim.x + threadIdx.x;
    if (p >= KDIM * NGATE) return;
    int e    = p & 15;
    int lane = (p >> 4) & 31;
    int nt   = (p >> 9) & 63;
    int kt   = p >> 15;
    int col  = nt * 16 + (lane & 15);
    int k    = kt * 32 + (lane >> 4) * 16 + e;
    float v  = (k < IDIM) ? W[(size_t)k * NGATE + col]
                          : U[(size_t)(k - IDIM) * NGATE + col];
    wpack[p] = f32_to_bf16(v);
}

// =====================================================================
// Kernel 1: fused LSTM. 16 blocks (batch tiles of M=16), 512 thr = 16 waves.
// Wave w owns gate columns d in [w*16, w*16+16) for ALL four gates ->
// gate math is lane-local; c lives in registers; h in the LDS A-tile.
// x_t tiles are staged via the Tensor Data Mover one step ahead.
// =====================================================================
__launch_bounds__(512, 1)
__global__ void lstm_fused(const float* __restrict__ x,      // (B,T,I) f32
                           const float* __restrict__ bias,   // (4D)   f32
                           const unsigned short* __restrict__ wpack,
                           float* __restrict__ out) {        // (B,2D) f32
    __shared__ unsigned short Atile[16][KDIM];               // bf16 bits
#if USE_TDM
    __shared__ __align__(16) float Xstage[16][IDIM];         // TDM landing zone (f32)
#endif

    const int tid   = threadIdx.x;
    const int lane  = tid & 31;
    const int wv    = tid >> 5;          // wave id 0..15 == d-tile
    const int bb    = blockIdx.x;        // batch tile

    // zero the h region of the A tile
    for (int i = tid; i < 16 * DDIM; i += 512) {
        Atile[i >> 8][IDIM + (i & 255)] = 0;
    }

    const int nloc  = lane & 15;
    const int mbase = (lane >> 4) * 8;           // C-layout row base for this lane half
    const int d     = wv * 16 + nloc;            // gate-relative column 0..255

    const float bi = bias[0 * DDIM + d];
    const float bf = bias[1 * DDIM + d];
    const float bg = bias[2 * DDIM + d];
    const float bo = bias[3 * DDIM + d];

    const int nt_i = wv, nt_f = 16 + wv, nt_g = 32 + wv, nt_o = 48 + wv;

    v8f c_state;
    float h_last[8];
    #pragma unroll
    for (int j = 0; j < 8; ++j) { c_state[j] = 0.0f; h_last[j] = 0.0f; }

    // x-tile mapping: thread -> 8 consecutive floats of a row
    const int xrow = tid >> 5;
    const int xcol = (tid & 31) * 8;

    const int arow  = lane & 15;     // A-matrix M = lane&15
    const int khalf = lane >> 4;

#if USE_TDM
    const unsigned stage_off = (unsigned)(uintptr_t)(void*)&Xstage[0][0]; // LDS byte offset
    if (wv == 0) {   // one TDM issue per block (per-wave op, EXEC-ignored)
        tdm_load_xtile(x + (size_t)(bb * 16) * TSTEPS * IDIM, stage_off);
    }
#endif

    #pragma unroll 1
    for (int t = 0; t < TSTEPS; ++t) {
#if USE_TDM
        if (wv == 0) {
            __builtin_amdgcn_s_wait_tensorcnt(0);  // x_t staged
        }
        __syncthreads();   // staging visible + prior step's A reads/h stores done

        // ---- convert staged x_t (f32) -> A tile (bf16) ----
        {
            const f32x4* src = (const f32x4*)&Xstage[xrow][xcol];
            f32x4 a = src[0];
            f32x4 b = src[1];
#else
        __syncthreads();   // prior step's A reads done before overwrite
        {
            const f32x4* src = (const f32x4*)(x + ((size_t)(bb * 16 + xrow) * TSTEPS + t) * IDIM + xcol);
            f32x4 a = src[0];
            f32x4 b = src[1];
#endif
            unsigned short tmp[8];
            tmp[0] = f32_to_bf16(a.x); tmp[1] = f32_to_bf16(a.y);
            tmp[2] = f32_to_bf16(a.z); tmp[3] = f32_to_bf16(a.w);
            tmp[4] = f32_to_bf16(b.x); tmp[5] = f32_to_bf16(b.y);
            tmp[6] = f32_to_bf16(b.z); tmp[7] = f32_to_bf16(b.w);
            u32x4 pk;
            pk.x = (unsigned int)tmp[0] | ((unsigned int)tmp[1] << 16);
            pk.y = (unsigned int)tmp[2] | ((unsigned int)tmp[3] << 16);
            pk.z = (unsigned int)tmp[4] | ((unsigned int)tmp[5] << 16);
            pk.w = (unsigned int)tmp[6] | ((unsigned int)tmp[7] << 16);
            *(u32x4*)&Atile[xrow][xcol] = pk;
        }
        __syncthreads();   // A tile complete; staging reads done block-wide

#if USE_TDM
        if (wv == 0 && t + 1 < TSTEPS) {   // pipeline next x tile behind the GEMM
            tdm_load_xtile(x + ((size_t)(bb * 16) * TSTEPS + (t + 1)) * IDIM, stage_off);
        }
#endif

        // ---- z = [x_t|h] @ [W;U] + b for this wave's d-range, all 4 gates ----
        v8f acc_i, acc_f, acc_g, acc_o;
        #pragma unroll
        for (int j = 0; j < 8; ++j) {
            acc_i[j] = bi; acc_f[j] = bf; acc_g[j] = bg; acc_o[j] = bo;
        }

        // software-pipelined B stream: current / next
        v16bf bvi = load_bfrag_global(wpack, 0, nt_i, lane);
        v16bf bvf = load_bfrag_global(wpack, 0, nt_f, lane);
        v16bf bvg = load_bfrag_global(wpack, 0, nt_g, lane);
        v16bf bvo = load_bfrag_global(wpack, 0, nt_o, lane);

        #pragma unroll 2
        for (int kt = 0; kt < KTILES; ++kt) {
            Bf16x16Bits abits;
            abits.lo = *(const u32x4*)&Atile[arow][kt * 32 + khalf * 16];
            abits.hi = *(const u32x4*)&Atile[arow][kt * 32 + khalf * 16 + 8];
            v16bf av = __builtin_bit_cast(v16bf, abits);

            const int ktn = (kt + 1 < KTILES) ? (kt + 1) : kt;   // branchless pipeline
            __builtin_prefetch((const void*)((const u32x4*)wpack +
                ((size_t)((ktn * NTILES + nt_i) * 32 + lane)) * 2), 0, 3);
            v16bf nvi = load_bfrag_global(wpack, ktn, nt_i, lane);
            v16bf nvf = load_bfrag_global(wpack, ktn, nt_f, lane);
            v16bf nvg = load_bfrag_global(wpack, ktn, nt_g, lane);
            v16bf nvo = load_bfrag_global(wpack, ktn, nt_o, lane);

            acc_i = __builtin_amdgcn_wmma_f32_16x16x32_bf16(false, av, false, bvi,
                                                            (short)0, acc_i, false, false);
            acc_f = __builtin_amdgcn_wmma_f32_16x16x32_bf16(false, av, false, bvf,
                                                            (short)0, acc_f, false, false);
            acc_g = __builtin_amdgcn_wmma_f32_16x16x32_bf16(false, av, false, bvg,
                                                            (short)0, acc_g, false, false);
            acc_o = __builtin_amdgcn_wmma_f32_16x16x32_bf16(false, av, false, bvo,
                                                            (short)0, acc_o, false, false);

            bvi = nvi; bvf = nvf; bvg = nvg; bvo = nvo;
        }

        // ---- gate math, lane-local: element j is (m = mbase+j, col d) ----
        #pragma unroll
        for (int j = 0; j < 8; ++j) {
            float ig = hsig(acc_i[j]);
            float fg = hsig(acc_f[j]);
            float gg = tanh_f(acc_g[j]);
            float og = hsig(acc_o[j]);
            float cn = fmaf(fg, c_state[j], ig * gg);
            float hn = og * tanh_f(cn);
            c_state[j] = cn;
            h_last[j]  = hn;
            Atile[mbase + j][IDIM + d] = f32_to_bf16(hn);   // h for next step
        }
    }

    // ---- write final [h_T | c_T] ----
    #pragma unroll
    for (int j = 0; j < 8; ++j) {
        size_t row = (size_t)(bb * 16 + mbase + j);
        out[row * (2 * DDIM) + d]        = h_last[j];
        out[row * (2 * DDIM) + DDIM + d] = c_state[j];
    }
}

// =====================================================================
extern "C" void kernel_launch(void* const* d_in, const int* in_sizes, int n_in,
                              void* d_out, int out_size, void* d_ws, size_t ws_size,
                              hipStream_t stream) {
    const float* x = (const float*)d_in[0];   // (256,512,256)
    const float* W = (const float*)d_in[1];   // (256,1024)
    const float* U = (const float*)d_in[2];   // (256,1024)
    const float* b = (const float*)d_in[3];   // (1024)
    float* out = (float*)d_out;               // (256,512)

    unsigned short* wpack = (unsigned short*)d_ws;   // 1 MB packed bf16 [W;U]

    int total = KDIM * NGATE;
    lstm_pack_weights<<<(total + 255) / 256, 256, 0, stream>>>(W, U, wpack);
    lstm_fused<<<BATCH / 16, 512, 0, stream>>>(x, b, wpack, out);
}
